// Model_17867063952091
// MI455X (gfx1250) — compile-verified, matrix operand order checked by reading, not run on hardware
//
#include <hip/hip_runtime.h>
#include <math.h>

typedef __attribute__((ext_vector_type(16))) _Float16 v16h;
typedef __attribute__((ext_vector_type(8)))  float    v8f;
typedef __attribute__((ext_vector_type(4)))  unsigned int v4u;
typedef __attribute__((ext_vector_type(8)))  int    v8i;
typedef __attribute__((ext_vector_type(4)))  int    v4i;

// Tensor Data Mover path (gfx1250): this toolchain exposes the 6-arg builtin
// (uint32x4 g0, int32x8 g1, int32x4, int32x4, int32x8, i32 cpol).
#if defined(__HIP_DEVICE_COMPILE__) && defined(__has_builtin)
#if __has_builtin(__builtin_amdgcn_tensor_load_to_lds) && __has_builtin(__builtin_amdgcn_s_wait_tensorcnt)
#define USE_TDM 1
#endif
#endif

namespace {

constexpr int kLayers = 12;
constexpr int kDModel = 512;
constexpr int kNHead  = 8;
constexpr int kDHead  = 64;
constexpr int kDFF    = 2048;
constexpr int kMem    = 512;
constexpr int kTok    = 308;
constexpr int kQ      = 512;
constexpr int kB      = 4;
constexpr int kK      = 1024;    // klen = q + mem

constexpr int kChunkK   = 256;   // K-chunk staged through LDS by the TDM
constexpr int kLdsPitch = 260;   // 256 + 4 dword pad per row (bank spread)

// Load 8 contiguous fp32, optionally add 8 fp32 bias, convert to f16 halves.
template<bool V, int OFF>
__device__ __forceinline__ void load8(v16h& d, const float* p, const float* va) {
  const float4* p4 = reinterpret_cast<const float4*>(p);
  float4 x0 = p4[0];
  float4 x1 = p4[1];
  float f0 = x0.x, f1 = x0.y, f2 = x0.z, f3 = x0.w;
  float f4 = x1.x, f5 = x1.y, f6 = x1.z, f7 = x1.w;
  if (V) {
    const float4* v4 = reinterpret_cast<const float4*>(va);
    float4 y0 = v4[0];
    float4 y1 = v4[1];
    f0 += y0.x; f1 += y0.y; f2 += y0.z; f3 += y0.w;
    f4 += y1.x; f5 += y1.y; f6 += y1.z; f7 += y1.w;
  }
  d[OFF + 0] = (_Float16)f0; d[OFF + 1] = (_Float16)f1;
  d[OFF + 2] = (_Float16)f2; d[OFF + 3] = (_Float16)f3;
  d[OFF + 4] = (_Float16)f4; d[OFF + 5] = (_Float16)f5;
  d[OFF + 6] = (_Float16)f6; d[OFF + 7] = (_Float16)f7;
}

// ---------------------------------------------------------------------------
// Register-direct batched GEMM (attention path, K=64/1024, strided operands).
// 2x2 blocking: each wave owns a 32x32 output block = 4 WMMA tiles; A frags
// are reused across the N pair and B frags across the M pair.
// ---------------------------------------------------------------------------
struct GemmP {
  const float* A;
  const float* B;
  float*       C;
  const float* vadd;   // length K, added to each A row (u/v rel-attn bias)
  int lda, ldb, ldc;
  int Mt, Nt, Nreal, K;   // Mt/Nt in 32-wide units
  int mod;                // batch decomposition: z -> (z/mod, z%mod)
  long aOuter, aInner, bOuter, bInner, cOuter, cInner;
  int vInner;
};

template<bool BT, bool HASVEC>
__global__ void wmma_gemm_kernel(GemmP p) {
  const int wave = threadIdx.x >> 5;
  const int lane = threadIdx.x & 31;
  const int tile = blockIdx.x * (blockDim.x >> 5) + wave;
  if (tile >= p.Mt * p.Nt) return;                  // wave-uniform guard
  const int tn = tile % p.Nt;
  const int tm = tile / p.Nt;
  const int z  = blockIdx.z;
  const int zo = z / p.mod;
  const int zi = z - zo * p.mod;
  const float* Ab = p.A + zo * p.aOuter + zi * p.aInner;
  const float* Bb = p.B + zo * p.bOuter + zi * p.bInner;
  float*       Cb = p.C + zo * p.cOuter + zi * p.cInner;
  const float* va = HASVEC ? (p.vadd + (long)zi * p.vInner) : nullptr;

  const int l16 = lane & 15;
  const int hi  = lane >> 4;                        // half-wave: K phase
  const int arow0 = tm * 32 + l16;
  const int arow1 = arow0 + 16;
  int bcol0 = tn * 32 + l16;
  int bcol1 = bcol0 + 16;
  if (bcol0 > p.Nreal - 1) bcol0 = p.Nreal - 1;     // branchless pad clamp
  if (bcol1 > p.Nreal - 1) bcol1 = p.Nreal - 1;
  const float* Arow0 = Ab + (long)arow0 * p.lda;
  const float* Arow1 = Ab + (long)arow1 * p.lda;
  const float* Brow0 = Bb + (long)bcol0 * p.ldb;    // BT only
  const float* Brow1 = Bb + (long)bcol1 * p.ldb;

  v8f c00 = {0.f,0.f,0.f,0.f,0.f,0.f,0.f,0.f};
  v8f c01 = c00, c10 = c00, c11 = c00;
  for (int k0 = 0; k0 < p.K; k0 += 32) {
    const int kA = k0 + hi * 8;                     // ISA 16-bit A layout
    v16h a0, a1, b0, b1;
    load8<HASVEC, 0>(a0, Arow0 + kA,      HASVEC ? va + kA      : nullptr);
    load8<HASVEC, 8>(a0, Arow0 + kA + 16, HASVEC ? va + kA + 16 : nullptr);
    load8<HASVEC, 0>(a1, Arow1 + kA,      HASVEC ? va + kA      : nullptr);
    load8<HASVEC, 8>(a1, Arow1 + kA + 16, HASVEC ? va + kA + 16 : nullptr);
    if (BT) {
      load8<false, 0>(b0, Brow0 + kA,      nullptr);
      load8<false, 8>(b0, Brow0 + kA + 16, nullptr);
      load8<false, 0>(b1, Brow1 + kA,      nullptr);
      load8<false, 8>(b1, Brow1 + kA + 16, nullptr);
      __builtin_prefetch(Arow0 + kA + 32, 0, 1);    // global_prefetch_b8
      __builtin_prefetch(Brow0 + kA + 32, 0, 1);
    } else {
      #pragma unroll
      for (int t = 0; t < 8; ++t) {
        b0[t]     = (_Float16)Bb[(long)(kA + t)      * p.ldb + bcol0];
        b0[8 + t] = (_Float16)Bb[(long)(kA + 16 + t) * p.ldb + bcol0];
        b1[t]     = (_Float16)Bb[(long)(kA + t)      * p.ldb + bcol1];
        b1[8 + t] = (_Float16)Bb[(long)(kA + 16 + t) * p.ldb + bcol1];
      }
    }
    c00 = __builtin_amdgcn_wmma_f32_16x16x32_f16(false, a0, false, b0, (short)0, c00, false, false);
    c01 = __builtin_amdgcn_wmma_f32_16x16x32_f16(false, a0, false, b1, (short)0, c01, false, false);
    c10 = __builtin_amdgcn_wmma_f32_16x16x32_f16(false, a1, false, b0, (short)0, c10, false, false);
    c11 = __builtin_amdgcn_wmma_f32_16x16x32_f16(false, a1, false, b1, (short)0, c11, false, false);
  }

  const int cc0 = tn * 32 + l16;
  const int cc1 = cc0 + 16;
  #pragma unroll
  for (int j = 0; j < 8; ++j) {
    const int r0 = tm * 32 + hi * 8 + j;            // ISA f32 C/D layout
    const int r1 = r0 + 16;
    if (cc0 < p.Nreal) {
      Cb[(long)r0 * p.ldc + cc0] = c00[j];
      Cb[(long)r1 * p.ldc + cc0] = c10[j];
    }
    if (cc1 < p.Nreal) {
      Cb[(long)r0 * p.ldc + cc1] = c01[j];
      Cb[(long)r1 * p.ldc + cc1] = c11[j];
    }
  }
}

// ---------------------------------------------------------------------------
// TDM-staged GEMM (weight GEMMs: C = A @ B^T [+bias][relu][+=C]).
// Block = 32 output rows x 8 waves x 32 cols. The shared 32-row A panel is
// DMA'd into LDS in 256-col chunks by the Tensor Data Mover (wave 0 issues a
// 2D D#; pad feature gives a 260-dword LDS pitch), synchronized with
// s_wait_tensorcnt + workgroup barrier. B streams from L2 (weights resident).
// ---------------------------------------------------------------------------
struct GemmT {
  const float* A;
  const float* B;
  float*       C;
  const float* bias;
  int lda, ldb, ldc;
  int Nt, Nreal, K;       // Nt in 32-wide units
};

template<bool RELU, bool ACCUM, bool HASBIAS>
__global__ void wmma_gemm_tdm_kernel(GemmT p) {
  __shared__ float As[32 * kLdsPitch];
  const int wave = threadIdx.x >> 5;
  const int lane = threadIdx.x & 31;
  const int l16  = lane & 15;
  const int hi   = lane >> 4;
  const int tn   = blockIdx.x * 8 + wave;
  const bool active = tn < p.Nt;                    // wave-uniform; no early return (barriers)
  const int mBase = blockIdx.y * 32;

  int bcol0 = tn * 32 + l16;
  int bcol1 = bcol0 + 16;
  if (bcol0 > p.Nreal - 1) bcol0 = p.Nreal - 1;
  if (bcol1 > p.Nreal - 1) bcol1 = p.Nreal - 1;
  const float* Brow0 = p.B + (long)bcol0 * p.ldb;
  const float* Brow1 = p.B + (long)bcol1 * p.ldb;
  const float* Apanel = p.A + (long)mBase * p.lda;

  v8f c00 = {0.f,0.f,0.f,0.f,0.f,0.f,0.f,0.f};
  v8f c01 = c00, c10 = c00, c11 = c00;

  for (int kc = 0; kc < p.K; kc += kChunkK) {
#ifdef USE_TDM
    if (wave == 0) {
      // D# group0: count=1 | lds_addr | global_addr | type=2 ("image")
      unsigned long long ga =
          (unsigned long long)(const void*)(Apanel + kc);
      unsigned lds = (unsigned)(unsigned long long)(__SIZE_TYPE__)&As[0];
      v4u g0 = { 1u, lds,
                 (unsigned)(ga & 0xffffffffull),
                 (unsigned)((ga >> 32) & 0x1ffffffull) | (2u << 30) };
      // D# group1: data_size=4B, pad_enable, pad_interval=256dw, pad_amount=4dw,
      // tensor_dim0=256, tensor_dim1=32, tile_dim0=256, tile_dim1=32,
      // tensor_dim0_stride=lda (elements)
      v8i g1 = { (int)0x07d20000u, (int)(256u << 16), (int)(32u << 16),
                 (int)(256u << 16), 32, p.lda, 0, 0 };
      v4i gz4 = { 0, 0, 0, 0 };
      v8i gz8 = { 0, 0, 0, 0, 0, 0, 0, 0 };
      __builtin_amdgcn_tensor_load_to_lds(g0, g1, gz4, gz4, gz8, 0);
      __builtin_amdgcn_s_wait_tensorcnt(0);
    }
#else
    // Fallback: cooperative vector copy global -> LDS (same 260-dw pitch).
    for (int t = threadIdx.x; t < 32 * (kChunkK / 4); t += 256) {
      const int r  = t >> 6;            // kChunkK/4 = 64 float4 per row
      const int c4 = t & 63;
      reinterpret_cast<float4*>(As + r * kLdsPitch)[c4] =
          reinterpret_cast<const float4*>(Apanel + (long)r * p.lda + kc)[c4];
    }
#endif
    __syncthreads();

    if (active) {
      for (int kk = 0; kk < kChunkK; kk += 32) {
        const int kl = kk + hi * 8;
        v16h a0, a1, b0, b1;
        load8<false, 0>(a0, As + l16 * kLdsPitch + kl,              nullptr);
        load8<false, 8>(a0, As + l16 * kLdsPitch + kl + 16,         nullptr);
        load8<false, 0>(a1, As + (l16 + 16) * kLdsPitch + kl,       nullptr);
        load8<false, 8>(a1, As + (l16 + 16) * kLdsPitch + kl + 16,  nullptr);
        load8<false, 0>(b0, Brow0 + kc + kl,       nullptr);
        load8<false, 8>(b0, Brow0 + kc + kl + 16,  nullptr);
        load8<false, 0>(b1, Brow1 + kc + kl,       nullptr);
        load8<false, 8>(b1, Brow1 + kc + kl + 16,  nullptr);
        __builtin_prefetch(Brow0 + kc + kl + 32, 0, 1);
        __builtin_prefetch(Brow1 + kc + kl + 32, 0, 1);
        c00 = __builtin_amdgcn_wmma_f32_16x16x32_f16(false, a0, false, b0, (short)0, c00, false, false);
        c01 = __builtin_amdgcn_wmma_f32_16x16x32_f16(false, a0, false, b1, (short)0, c01, false, false);
        c10 = __builtin_amdgcn_wmma_f32_16x16x32_f16(false, a1, false, b0, (short)0, c10, false, false);
        c11 = __builtin_amdgcn_wmma_f32_16x16x32_f16(false, a1, false, b1, (short)0, c11, false, false);
      }
    }
    __syncthreads();
  }

  if (active) {
    const int cc0 = tn * 32 + l16;
    const int cc1 = cc0 + 16;
    const float bia0 = HASBIAS ? p.bias[cc0] : 0.f;
    const float bia1 = HASBIAS ? p.bias[cc1 < p.Nreal ? cc1 : p.Nreal - 1] : 0.f;
    #pragma unroll
    for (int j = 0; j < 8; ++j) {
      const int r0 = mBase + hi * 8 + j;
      const int r1 = r0 + 16;
      if (cc0 < p.Nreal) {
        float* p0 = p.C + (long)r0 * p.ldc + cc0;
        float* p1 = p.C + (long)r1 * p.ldc + cc0;
        float v0 = c00[j] + bia0, v1 = c10[j] + bia0;
        if (RELU)  { v0 = v0 > 0.f ? v0 : 0.f; v1 = v1 > 0.f ? v1 : 0.f; }
        if (ACCUM) { v0 += *p0; v1 += *p1; }
        *p0 = v0; *p1 = v1;
      }
      if (cc1 < p.Nreal) {
        float* p0 = p.C + (long)r0 * p.ldc + cc1;
        float* p1 = p.C + (long)r1 * p.ldc + cc1;
        float v0 = c01[j] + bia1, v1 = c11[j] + bia1;
        if (RELU)  { v0 = v0 > 0.f ? v0 : 0.f; v1 = v1 > 0.f ? v1 : 0.f; }
        if (ACCUM) { v0 += *p0; v1 += *p1; }
        *p0 = v0; *p1 = v1;
      }
    }
  }
}

// cat = [zeros(mlen,b,d) ; word_emb[x].T(q,b,d)]  (mems stay zero all layers)
__global__ void embed_cat_kernel(const int* x, const float* emb, float* cat) {
  int idx = blockIdx.x * blockDim.x + threadIdx.x;
  const int total = kK * kB * kDModel;
  if (idx >= total) return;
  int c  = idx & (kDModel - 1);
  int rb = idx >> 9;
  int bb = rb & (kB - 1);
  int j  = rb >> 2;
  float v = 0.f;
  if (j >= kMem) {
    int tok = x[bb * kQ + (j - kMem)];
    v = emb[tok * kDModel + c];
  }
  cat[idx] = v;
}

// Sinusoidal relative positions, pos_seq = klen-1 .. 0
__global__ void pos_emb_kernel(float* r) {
  int idx = blockIdx.x * blockDim.x + threadIdx.x;
  const int total = kK * kDModel;
  if (idx >= total) return;
  int c = idx & (kDModel - 1);
  int j = idx >> 9;
  float pos = (float)(kK - 1 - j);
  int ci = (c < 256) ? c : (c - 256);
  float invf = __powf(10000.f, -(float)ci * (1.f / 256.f));
  float ang = pos * invf;
  r[idx] = (c < 256) ? __sinf(ang) : __cosf(ang);
}

// One wave per 512-wide row; biased variance; eps = 1e-3
__global__ void layernorm_kernel(const float* in, float* out,
                                 const float* g, const float* b, int nrows) {
  int wave = (blockIdx.x * blockDim.x + threadIdx.x) >> 5;
  int lane = threadIdx.x & 31;
  if (wave >= nrows) return;
  const float* row = in + (long)wave * kDModel;
  float* orow = out + (long)wave * kDModel;
  float v[16];
  float s = 0.f;
  #pragma unroll
  for (int t = 0; t < 16; ++t) { v[t] = row[t * 32 + lane]; s += v[t]; }
  #pragma unroll
  for (int o = 16; o > 0; o >>= 1) s += __shfl_xor(s, o, 32);
  float mu = s * (1.f / kDModel);
  float q = 0.f;
  #pragma unroll
  for (int t = 0; t < 16; ++t) { float d = v[t] - mu; q += d * d; }
  #pragma unroll
  for (int o = 16; o > 0; o >>= 1) q += __shfl_xor(q, o, 32);
  float inv = rsqrtf(q * (1.f / kDModel) + 1e-3f);
  #pragma unroll
  for (int t = 0; t < 16; ++t) {
    int c = t * 32 + lane;
    orow[c] = (v[t] - mu) * inv * g[c] + b[c];
  }
}

// score = (AC + rel_shift(BD)) * scale, causal mask, softmax over klen.
// In-place over AC -> prob. One wave per (z=b*8+h, i) row of 1024.
__global__ void score_softmax_kernel(float* ac, const float* bd) {
  int row  = (blockIdx.x * blockDim.x + threadIdx.x) >> 5;
  int lane = threadIdx.x & 31;
  if (row >= 32 * kQ) return;
  int z = row >> 9;
  int i = row & (kQ - 1);
  float* arow = ac + (long)z * kQ * kK + (long)i * kK;
  const float* bz = bd + (long)z * kQ * kK;
  const float scale = 0.125f;  // 1/sqrt(64)
  float vals[32];
  float mx = -3.0e38f;
  #pragma unroll
  for (int t = 0; t < 32; ++t) {
    int j  = t * 32 + lane;
    int tt = i * kK + j + kQ;              // rel_shift flat index (+q)
    int i0 = tt / (kK + 1);
    int j0 = tt - i0 * (kK + 1);
    float bdv = (j0 == 0) ? 0.f : bz[i0 * kK + (j0 - 1)];
    float s = (arow[j] + bdv) * scale;
    if (j > i + kMem) s = -1e30f;          // causal mask with mem offset
    vals[t] = s;
    mx = fmaxf(mx, s);
  }
  #pragma unroll
  for (int o = 16; o > 0; o >>= 1) mx = fmaxf(mx, __shfl_xor(mx, o, 32));
  float sum = 0.f;
  #pragma unroll
  for (int t = 0; t < 32; ++t) { vals[t] = __expf(vals[t] - mx); sum += vals[t]; }
  #pragma unroll
  for (int o = 16; o > 0; o >>= 1) sum += __shfl_xor(sum, o, 32);
  float inv = 1.f / sum;
  #pragma unroll
  for (int t = 0; t < 32; ++t) arow[t * 32 + lane] = vals[t] * inv;
}

// h (q,b,d) time-major -> ht (b,q,d) batch-major for the output head
__global__ void transpose_h_kernel(const float* h, float* ht) {
  int idx = blockIdx.x * blockDim.x + threadIdx.x;
  const int total = kQ * kB * kDModel;
  if (idx >= total) return;
  int c  = idx & (kDModel - 1);
  int rb = idx >> 9;
  int bb = rb & 3;
  int i  = rb >> 2;
  ht[((long)(bb * kQ + i)) * kDModel + c] = h[idx];
}

}  // namespace

extern "C" void kernel_launch(void* const* d_in, const int* in_sizes, int n_in,
                              void* d_out, int out_size, void* d_ws, size_t ws_size,
                              hipStream_t stream) {
  (void)in_sizes; (void)n_in; (void)out_size; (void)ws_size;
  const int*   x     = (const int*)  d_in[0];
  const float* wemb  = (const float*)d_in[1];
  const float* qkv_w = (const float*)d_in[2];
  const float* r_w   = (const float*)d_in[3];
  const float* o_w   = (const float*)d_in[4];
  const float* ln1_g = (const float*)d_in[5];
  const float* ln1_b = (const float*)d_in[6];
  const float* ff_w1 = (const float*)d_in[7];
  const float* ff_b1 = (const float*)d_in[8];
  const float* ff_w2 = (const float*)d_in[9];
  const float* ff_b2 = (const float*)d_in[10];
  const float* ln2_g = (const float*)d_in[11];
  const float* ln2_b = (const float*)d_in[12];
  const float* u     = (const float*)d_in[13];
  const float* v     = (const float*)d_in[14];
  const float* out_w = (const float*)d_in[15];
  const float* out_b = (const float*)d_in[16];
  float* out = (float*)d_out;

  float* ws = (float*)d_ws;
  size_t off = 0;
  auto wsalloc = [&](size_t n) { float* p = ws + off; off += n; return p; };
  float* cat   = wsalloc((size_t)kK * kB * kDModel);       // (klen,b,d)
  float* cat_n = wsalloc((size_t)kK * kB * kDModel);
  float* rbuf  = wsalloc((size_t)kK * kDModel);
  float* rk    = wsalloc((size_t)kK * kDModel);            // (klen, h*dh)
  float* heads = wsalloc((size_t)kK * kB * 3 * kDModel);   // q|k|v
  float* ac    = wsalloc((size_t)32 * kQ * kK);            // scores / prob
  float* bdbuf = wsalloc((size_t)32 * kQ * kK);
  float* vec   = wsalloc((size_t)kQ * kB * kDModel);
  float* hn    = wsalloc((size_t)kQ * kB * kDModel);
  float* ff1   = wsalloc((size_t)kQ * kB * kDFF);
  float* ht    = wsalloc((size_t)kQ * kB * kDModel);
  float* h = cat + (size_t)kMem * kB * kDModel;            // live state in cat tail

  const int HD3 = 3 * kDModel;                             // 1536

  auto tdm_grid = [](int M, int Nt) {
    return dim3((unsigned)((Nt + 7) / 8), (unsigned)(M / 32), 1);
  };

  embed_cat_kernel<<<(kK * kB * kDModel + 255) / 256, 256, 0, stream>>>(x, wemb, cat);
  pos_emb_kernel<<<(kK * kDModel + 255) / 256, 256, 0, stream>>>(rbuf);

  for (int l = 0; l < kLayers; ++l) {
    // pre-LN over [mems; h]
    layernorm_kernel<<<(kK * kB + 7) / 8, 256, 0, stream>>>(
        cat, cat_n, ln1_g + (size_t)l * kDModel, ln1_b + (size_t)l * kDModel, kK * kB);

    // heads = cat_n @ qkv_w[l].T   (4096 x 512 x 1536)
    {
      GemmT p{}; p.A = cat_n; p.lda = kDModel;
      p.B = qkv_w + (size_t)l * HD3 * kDModel; p.ldb = kDModel;
      p.C = heads; p.ldc = HD3;
      p.Nt = HD3 / 32; p.Nreal = HD3; p.K = kDModel;
      wmma_gemm_tdm_kernel<false, false, false>
          <<<tdm_grid(kK * kB, p.Nt), 256, 0, stream>>>(p);
    }
    // rk = r @ r_w[l].T   (1024 x 512 x 512)
    {
      GemmT p{}; p.A = rbuf; p.lda = kDModel;
      p.B = r_w + (size_t)l * kDModel * kDModel; p.ldb = kDModel;
      p.C = rk; p.ldc = kDModel;
      p.Nt = kDModel / 32; p.Nreal = kDModel; p.K = kDModel;
      wmma_gemm_tdm_kernel<false, false, false>
          <<<tdm_grid(kK, p.Nt), 256, 0, stream>>>(p);
    }
    // AC[z] = (q + u) @ k^T per (b,head): z = bb*8 + n, 512 x 64 x 1024
    {
      GemmP p{}; p.A = heads + (size_t)kMem * kB * HD3; p.lda = kB * HD3;
      p.B = heads + kDModel; p.ldb = kB * HD3;
      p.C = ac; p.ldc = kK;
      p.Mt = kQ / 32; p.Nt = kK / 32; p.Nreal = kK; p.K = kDHead; p.mod = kNHead;
      p.aOuter = HD3; p.aInner = kDHead;
      p.bOuter = HD3; p.bInner = kDHead;
      p.cOuter = (long)kNHead * kQ * kK; p.cInner = (long)kQ * kK;
      p.vadd = u; p.vInner = kDHead;
      dim3 g((unsigned)((p.Mt * p.Nt + 7) / 8), 1, kB * kNHead);
      wmma_gemm_kernel<true, true><<<g, 256, 0, stream>>>(p);
    }
    // BD[z] = (q + v) @ rk[:,n,:]^T
    {
      GemmP p{}; p.A = heads + (size_t)kMem * kB * HD3; p.lda = kB * HD3;
      p.B = rk; p.ldb = kDModel;
      p.C = bdbuf; p.ldc = kK;
      p.Mt = kQ / 32; p.Nt = kK / 32; p.Nreal = kK; p.K = kDHead; p.mod = kNHead;
      p.aOuter = HD3; p.aInner = kDHead;
      p.bOuter = 0;   p.bInner = kDHead;
      p.cOuter = (long)kNHead * kQ * kK; p.cInner = (long)kQ * kK;
      p.vadd = v; p.vInner = kDHead;
      dim3 g((unsigned)((p.Mt * p.Nt + 7) / 8), 1, kB * kNHead);
      wmma_gemm_kernel<true, true><<<g, 256, 0, stream>>>(p);
    }
    // fused rel_shift + mask + scale + softmax (prob in place over ac)
    score_softmax_kernel<<<(32 * kQ) / 8, 256, 0, stream>>>(ac, bdbuf);
    // vec[z] = prob @ V   (B not transposed: strided K)
    {
      GemmP p{}; p.A = ac; p.lda = kK;
      p.B = heads + 2 * kDModel; p.ldb = kB * HD3;
      p.C = vec; p.ldc = kB * kDModel;
      p.Mt = kQ / 32; p.Nt = kDHead / 32; p.Nreal = kDHead; p.K = kK; p.mod = kNHead;
      p.aOuter = (long)kNHead * kQ * kK; p.aInner = (long)kQ * kK;
      p.bOuter = HD3; p.bInner = kDHead;
      p.cOuter = kDModel; p.cInner = kDHead;
      dim3 g((unsigned)((p.Mt * p.Nt + 7) / 8), 1, kB * kNHead);
      wmma_gemm_kernel<false, false><<<g, 256, 0, stream>>>(p);
    }
    // h += vec @ o_w[l].T  (residual fused via accumulate)
    {
      GemmT p{}; p.A = vec; p.lda = kDModel;
      p.B = o_w + (size_t)l * kDModel * kDModel; p.ldb = kDModel;
      p.C = h; p.ldc = kDModel;
      p.Nt = kDModel / 32; p.Nreal = kDModel; p.K = kDModel;
      wmma_gemm_tdm_kernel<false, true, false>
          <<<tdm_grid(kQ * kB, p.Nt), 256, 0, stream>>>(p);
    }
    // FF pre-LN
    layernorm_kernel<<<(kQ * kB + 7) / 8, 256, 0, stream>>>(
        h, hn, ln2_g + (size_t)l * kDModel, ln2_b + (size_t)l * kDModel, kQ * kB);
    // ff1 = relu(hn @ ff_w1.T + b1)
    {
      GemmT p{}; p.A = hn; p.lda = kDModel;
      p.B = ff_w1 + (size_t)l * kDFF * kDModel; p.ldb = kDModel;
      p.C = ff1; p.ldc = kDFF;
      p.bias = ff_b1 + (size_t)l * kDFF;
      p.Nt = kDFF / 32; p.Nreal = kDFF; p.K = kDModel;
      wmma_gemm_tdm_kernel<true, false, true>
          <<<tdm_grid(kQ * kB, p.Nt), 256, 0, stream>>>(p);
    }
    // h += ff1 @ ff_w2.T + b2
    {
      GemmT p{}; p.A = ff1; p.lda = kDFF;
      p.B = ff_w2 + (size_t)l * kDModel * kDFF; p.ldb = kDFF;
      p.C = h; p.ldc = kDModel;
      p.bias = ff_b2 + (size_t)l * kDModel;
      p.Nt = kDModel / 32; p.Nreal = kDModel; p.K = kDFF;
      wmma_gemm_tdm_kernel<false, true, true>
          <<<tdm_grid(kQ * kB, p.Nt), 256, 0, stream>>>(p);
    }
  }

  // output head: out = ht @ out_w.T + out_b  (N=308, padded tiles guarded)
  transpose_h_kernel<<<(kQ * kB * kDModel + 255) / 256, 256, 0, stream>>>(h, ht);
  {
    GemmT p{}; p.A = ht; p.lda = kDModel;
    p.B = out_w; p.ldb = kDModel;
    p.C = out; p.ldc = kTok;
    p.bias = out_b;
    p.Nt = (kTok + 31) / 32; p.Nreal = kTok; p.K = kDModel;
    wmma_gemm_tdm_kernel<false, false, true>
        <<<tdm_grid(kQ * kB, p.Nt), 256, 0, stream>>>(p);
  }
}
